// CliffordLinear_31817117728926
// MI455X (gfx1250) — matrix operand order repaired and checked by reading, not compiled, not run on hardware
//
#include <hip/hip_runtime.h>

// ---------------------------------------------------------------------------
// CliffordLinear as a single 65536 x 512 x 512 bf16-WMMA GEMM for gfx1250.
//   out[b, o*8+kk] = sum_k x[b,k] * WbigT[o*8+kk][k] + bias[o,kk] (+shift at kk==0)
//   k = n*8+i ; WbigT[col][k] = weight[o,n,j(i,kk)] * sign(i,kk) * rs[o]*cs[n]
// B tile moves global->LDS via GLOBAL_LOAD_ASYNC_TO_LDS_B128 (ASYNCcnt),
// double-buffered; A tile converts f32->bf16 in registers.
// ---------------------------------------------------------------------------

typedef __attribute__((ext_vector_type(16))) __bf16 v16bf;
typedef __attribute__((ext_vector_type(8)))  float  v8f;
typedef int v4i __attribute__((vector_size(16)));   // matches builtin's V4i

#define MROWS   65536
#define KDIM    512
#define NDIM    512
#define MTILE   64          // M rows per block
#define KCHUNK  32          // K per bf16 WMMA
#define NCHUNKS (KDIM / KCHUNK)
#define LDA     40          // LDS row stride (bf16 elems): 80B, 16B aligned
#define LDB     40
#define THREADS 512         // 16 wave32

#if __has_builtin(__builtin_amdgcn_global_load_async_to_lds_b128) && \
    __has_builtin(__builtin_amdgcn_s_wait_asynccnt)
#define USE_ASYNC_LDS 1
typedef __attribute__((address_space(1))) v4i GI4;  // global int4 (prints as __device__)
typedef __attribute__((address_space(3))) v4i LI4;  // LDS int4
#else
#define USE_ASYNC_LDS 0
#endif

// For each (input blade i, output blade kk): the unique j with C[i,j,kk]!=0, and sign.
__constant__ int c_jidx[64] = {
  0,1,2,3,4,5,6,7,
  1,0,4,5,2,3,7,6,
  2,4,0,6,1,7,3,5,
  3,5,6,0,7,1,2,4,
  4,2,1,7,0,6,5,3,
  5,3,7,1,6,0,4,2,
  6,7,3,2,5,4,0,1,
  7,6,5,4,3,2,1,0
};
__constant__ float c_sgn[64] = {
   1, 1, 1, 1, 1, 1, 1, 1,
   1, 1, 1, 1, 1, 1, 1, 1,
   1,-1, 1, 1,-1,-1, 1,-1,
   1,-1,-1, 1, 1,-1,-1, 1,
  -1, 1,-1,-1, 1, 1,-1, 1,
  -1, 1, 1,-1,-1, 1, 1,-1,
  -1,-1, 1,-1, 1,-1, 1, 1,
  -1,-1, 1,-1, 1,-1, 1, 1
};

__device__ __forceinline__ unsigned int f2bf(float f) {
  unsigned int u = __builtin_bit_cast(unsigned int, f);
  u += 0x7FFFu + ((u >> 16) & 1u);          // round-to-nearest-even
  return u >> 16;
}

union FragBF { v16bf v; uint4 u[2]; };

// ---------------------------------------------------------------------------
// Prep: build WbigT[col][k] in bf16 (col-major in k -> contiguous B columns).
// ---------------------------------------------------------------------------
__global__ void clifford_prep_kernel(const float* __restrict__ weight,
                                     const float* __restrict__ row_scale,
                                     const float* __restrict__ col_scale,
                                     unsigned short* __restrict__ WT) {
  int idx = blockIdx.x * blockDim.x + threadIdx.x;   // == col*512 + k
  int col = idx >> 9;
  int k   = idx & 511;
  int o = col >> 3, kk = col & 7;
  int n = k   >> 3, i  = k   & 7;
  int   j = c_jidx[i * 8 + kk];
  float s = c_sgn [i * 8 + kk];
  float wv = weight[(o * 64 + n) * 8 + j] * row_scale[o] * col_scale[n];
  if (wv != wv) wv = 0.0f;                            // nan_to_num
  WT[idx] = (unsigned short)f2bf(wv * s);
}

// ---------------------------------------------------------------------------
// GEMM: block = 64(M) x 512(N), 512 threads = 16 wave32.
// wave (mrow = wave&3, nq = wave>>2) owns a 16(M) x 128(N) strip
// = 8 WMMA tiles -> 64 f32 accumulator VGPRs.
// ---------------------------------------------------------------------------
__global__ void __launch_bounds__(THREADS)
clifford_gemm_kernel(const float* __restrict__ x,
                     const unsigned short* __restrict__ WT,
                     const float* __restrict__ bias,
                     const float* __restrict__ bias_shift,
                     float* __restrict__ out) {
  __shared__ __align__(16) unsigned short As[2][MTILE * LDA];  // 2 x  5 KB
  __shared__ __align__(16) unsigned short Bs[2][NDIM  * LDB];  // 2 x 40 KB

  const int tid   = threadIdx.x;
  const int lane  = tid & 31;
  const int wave  = tid >> 5;          // 0..15
  const int mrow  = wave & 3;          // 16-row strip of the 64
  const int nq    = wave >> 2;         // 128-col quarter
  const int Mbase = blockIdx.x * MTILE;

  v8f acc[8];
  #pragma unroll
  for (int t = 0; t < 8; ++t)
    #pragma unroll
    for (int e = 0; e < 8; ++e) acc[t][e] = 0.0f;

  // ---- A path: 64x32 f32 tile, 1 float4 per thread, f32->bf16 in regs ----
  const int arow = tid >> 3;           // 8 float4 pieces per row
  const int ac4  = tid & 7;
  float4 aReg;
  auto loadA = [&](int kc) {
    aReg = *(const float4*)(x + (Mbase + arow) * KDIM + kc * KCHUNK + ac4 * 4);
  };
  auto storeA = [&](int b) {
    uint2 v;
    v.x = f2bf(aReg.x) | (f2bf(aReg.y) << 16);
    v.y = f2bf(aReg.z) | (f2bf(aReg.w) << 16);
    *(uint2*)&As[b][arow * LDA + ac4 * 4] = v;
  };

  // ---- B path: 512 cols x 32 k bf16 tile = 2048 16B pieces, 4 per thread ----
#if USE_ASYNC_LDS
  auto issueB = [&](int kc, int b) {
    #pragma unroll
    for (int r = 0; r < 4; ++r) {
      int p   = tid + r * THREADS;
      int col = p >> 2;
      int seg = p & 3;
      __builtin_amdgcn_global_load_async_to_lds_b128(
          (GI4*)(WT + col * KDIM + kc * KCHUNK + seg * 8),
          (LI4*)&Bs[b][col * LDB + seg * 8],
          /*offset=*/0, /*cpol=*/0);
    }
  };
#else
  uint4 bReg[4];
  auto loadB = [&](int kc) {
    #pragma unroll
    for (int r = 0; r < 4; ++r) {
      int p   = tid + r * THREADS;
      int col = p >> 2;
      int seg = p & 3;
      bReg[r] = *(const uint4*)(WT + col * KDIM + kc * KCHUNK + seg * 8);
    }
  };
  auto storeB = [&](int b) {
    #pragma unroll
    for (int r = 0; r < 4; ++r) {
      int p   = tid + r * THREADS;
      int col = p >> 2;
      int seg = p & 3;
      *(uint4*)&Bs[b][col * LDB + seg * 8] = bReg[r];
    }
  };
#endif

  // Fragment addressing (ISA 16-bit A 16x32 / B 32x16 lane layouts)
  const int mloc  = mrow * 16 + (lane & 15);
  const int kb    = (lane >> 4) * 8;
  const int khalf = (lane >> 4) * 16;

  // ---- software pipeline, double-buffered LDS, one barrier per chunk ----
#if USE_ASYNC_LDS
  issueB(0, 0);
  loadA(0);
  int b = 0;
  for (int kc = 0; kc < NCHUNKS; ++kc) {
    storeA(b);
    __builtin_amdgcn_s_wait_asynccnt(0);   // chunk kc B resident in LDS
    __syncthreads();
    if (kc + 1 < NCHUNKS) { issueB(kc + 1, b ^ 1); loadA(kc + 1); }
#else
  loadB(0);
  loadA(0);
  int b = 0;
  for (int kc = 0; kc < NCHUNKS; ++kc) {
    storeA(b);
    storeB(b);
    __syncthreads();
    if (kc + 1 < NCHUNKS) { loadB(kc + 1); loadA(kc + 1); }
#endif
    FragBF fa;
    fa.u[0] = *(const uint4*)&As[b][mloc * LDA + kb];        // K = kb..kb+7
    fa.u[1] = *(const uint4*)&As[b][mloc * LDA + 16 + kb];   // K = 16+kb..
    #pragma unroll
    for (int nt = 0; nt < 8; ++nt) {
      int col = nq * 128 + nt * 16 + (lane & 15);
      FragBF fb;
      fb.u[0] = *(const uint4*)&Bs[b][col * LDB + kb];
      fb.u[1] = *(const uint4*)&Bs[b][col * LDB + 16 + kb];
      acc[nt] = __builtin_amdgcn_wmma_f32_16x16x32_bf16(
          false, fa.v, false, fb.v, (short)0, acc[nt], false, false);
    }
    b ^= 1;
  }

  // Epilogue: D layout — VGPR r: lanes 0-15 -> M=r, lanes 16-31 -> M=8+r; N = lane&15.
  const int baseRow = Mbase + mrow * 16 + (lane >> 4) * 8;
  #pragma unroll
  for (int nt = 0; nt < 8; ++nt) {
    int col = nq * 128 + nt * 16 + (lane & 15);
    float bval = bias[col];
    if ((col & 7) == 0) bval += bias_shift[col >> 3];
    #pragma unroll
    for (int r = 0; r < 8; ++r) {
      out[(baseRow + r) * NDIM + col] = acc[nt][r] + bval;
    }
  }
}

extern "C" void kernel_launch(void* const* d_in, const int* in_sizes, int n_in,
                              void* d_out, int out_size, void* d_ws, size_t ws_size,
                              hipStream_t stream) {
  const float* x          = (const float*)d_in[0];
  const float* weight     = (const float*)d_in[1];
  const float* bias       = (const float*)d_in[2];
  const float* row_scale  = (const float*)d_in[3];
  const float* col_scale  = (const float*)d_in[4];
  const float* bias_shift = (const float*)d_in[5];
  float* out              = (float*)d_out;
  unsigned short* WT      = (unsigned short*)d_ws;   // 512*512 bf16 = 512 KB

  clifford_prep_kernel<<<(KDIM * NDIM) / 256, 256, 0, stream>>>(
      weight, row_scale, col_scale, WT);
  clifford_gemm_kernel<<<MROWS / MTILE, THREADS, 0, stream>>>(
      x, WT, bias, bias_shift, out);
}
